// C2SModel_50620484550700
// MI455X (gfx1250) — matrix operand order repaired and checked by reading, not compiled, not run on hardware
//
#include <hip/hip_runtime.h>
#include <hip/hip_bf16.h>

// ---------------- model constants ----------------
#define NCTX   65536
#define BSEG   512
#define EMB    128     // E
#define HID    128     // H
#define TLEN   9
#define FOURH  512     // 4*H
#define CTXW   512     // 2E + 2H
#define DEC    320
#define OUTV   10000
#define NODE_V 512

#define LDSB_STRIDE 272   // padded row stride (elements) for the 64x256 LDS B tile

typedef __bf16 bf16_t;
typedef __attribute__((ext_vector_type(16))) __bf16 v16bf;
typedef __attribute__((ext_vector_type(8)))  float  v8f;

// ---------------- WMMA fragment helpers (contiguous k order, same for A and B) ----------------
// A-matrix 16x32 bf16: lanes 0-15 -> M=lane, K=0..15; lanes 16-31 -> K=16..31
__device__ __forceinline__ v16bf load_a_frag_row(const bf16_t* __restrict__ row, int k0, int lane) {
    int kb = k0 + ((lane & 16) ? 16 : 0);
    v16bf a;
#pragma unroll
    for (int j = 0; j < 8; ++j) {
        int k = kb + 2 * j;
        a[2 * j]     = row[k];
        a[2 * j + 1] = row[k + 1];
    }
    return a;
}
// B-matrix 32x16 bf16 (row of W[N,K]): lanes 0-15 -> K=0..15, lanes 16-31 -> K=16..31 (contig 32B)
__device__ __forceinline__ v16bf load_b_frag_row(const bf16_t* __restrict__ row, int k0, int lane) {
    int kb = k0 + ((lane & 16) ? 16 : 0);
    v16bf b;
#pragma unroll
    for (int j = 0; j < 8; ++j) {
        b[2 * j]     = row[kb + 2 * j];
        b[2 * j + 1] = row[kb + 2 * j + 1];
    }
    return b;
}

// ---------------- elementwise / setup kernels ----------------
__global__ void cast_f32_to_bf16(const float* __restrict__ src, bf16_t* __restrict__ dst, long n) {
    long i = (long)blockIdx.x * blockDim.x + threadIdx.x;
    if (i < n) dst[i] = (bf16_t)src[i];
}

__global__ void combine_bias(const float* __restrict__ a, const float* __restrict__ b,
                             float* __restrict__ out, int n) {
    int i = blockIdx.x * blockDim.x + threadIdx.x;
    if (i < n) out[i] = a[i] + b[i];
}

// scatter-add subtoken embeddings: one thread per (occurrence, 4-float chunk)
__global__ void seg_emb_add(const int* __restrict__ toks, const int* __restrict__ segs,
                            const float* __restrict__ emb, float* __restrict__ out, int L) {
    int tid = blockIdx.x * blockDim.x + threadIdx.x;
    const int per = EMB / 4;                 // 32 chunks per occurrence
    int occ = tid / per;
    if (occ >= L) return;
    int e = (tid % per) * 4;
    int tkn = toks[occ];
    int sg  = segs[occ];
    const float4 val = *(const float4*)(emb + (long)tkn * EMB + e);
    float* o = out + (long)sg * EMB + e;
    atomicAdd(o + 0, val.x);
    atomicAdd(o + 1, val.y);
    atomicAdd(o + 2, val.z);
    atomicAdd(o + 3, val.w);
}

// ---------------- LSTM step GEMM: g = x_t @ Wih^T + h @ Whh^T + bias ----------------
// grid = (NCTX/64, FOURH/64), block = 128 (4 waves; wave -> 16(M) x 64(N) tile).
// B tile (64 rows x 256 K = wih|whh concat) async-staged into LDS and shared by all 4 waves.
__global__ __launch_bounds__(128)
void lstm_gemm_step(const bf16_t* __restrict__ node_bf, const int* __restrict__ paths, int t,
                    const bf16_t* __restrict__ h_bf,
                    const bf16_t* __restrict__ wih, const bf16_t* __restrict__ whh,
                    const float* __restrict__ bias, float* __restrict__ g) {
    __shared__ bf16_t tileB[64 * LDSB_STRIDE];

    const int tid  = threadIdx.x;
    const int lane = tid & 31;
    const int wave = tid >> 5;
    const int n0 = blockIdx.y * 64;
    const int m0 = (blockIdx.x * 4 + wave) * 16;

    // ---- cooperative async stage of B tile into LDS (ASYNCcnt path) ----
    // 64 rows x 32 16-byte chunks = 2048 B128 transfers, 16 per thread
#pragma unroll
    for (int r = 0; r < 16; ++r) {
        int i     = tid + 128 * r;
        int row   = i >> 5;
        int chunk = i & 31;
        const bf16_t* src = (chunk < 16)
            ? (wih + (long)(n0 + row) * EMB + chunk * 8)
            : (whh + (long)(n0 + row) * HID + (chunk - 16) * 8);
        unsigned           ldsoff = (unsigned)(size_t)&tileB[row * LDSB_STRIDE + chunk * 8];
        unsigned long long gaddr  = (unsigned long long)(size_t)src;
        asm volatile("global_load_async_to_lds_b128 %0, %1, off"
                     :: "v"(ldsoff), "v"(gaddr) : "memory");
    }
    asm volatile("s_wait_asynccnt 0x0" ::: "memory");
    __syncthreads();

    const int m = m0 + (lane & 15);
    const bf16_t* arow_x = node_bf + (long)paths[(long)m * TLEN + t] * EMB;  // on-the-fly gather
    const bf16_t* arow_h = h_bf + (long)m * HID;

    v8f acc[4] = {};
#pragma unroll
    for (int k0 = 0; k0 < EMB + HID; k0 += 32) {
        const bf16_t* ar = (k0 < EMB) ? arow_x : arow_h;
        const int     kk = (k0 < EMB) ? k0 : (k0 - EMB);
        v16bf af = load_a_frag_row(ar, kk, lane);
#pragma unroll
        for (int u = 0; u < 4; ++u) {
            const bf16_t* br = tileB + (16 * u + (lane & 15)) * LDSB_STRIDE;
            v16bf bf = load_b_frag_row(br, k0, lane);
            acc[u] = __builtin_amdgcn_wmma_f32_16x16x32_bf16(false, af, false, bf,
                                                             (short)0, acc[u], false, false);
        }
    }

    const int mb = m0 + ((lane & 16) ? 8 : 0);
#pragma unroll
    for (int u = 0; u < 4; ++u) {
        const int n  = n0 + 16 * u + (lane & 15);
        const float bv = bias[n];
#pragma unroll
        for (int rr = 0; rr < 8; ++rr)
            g[(long)(mb + rr) * FOURH + n] = acc[u][rr] + bv;
    }
}

// gate nonlinearity + state update; PyTorch gate order i,f,g,o
__global__ void lstm_gate(const float* __restrict__ g, float* __restrict__ c,
                          float* __restrict__ h_out, bf16_t* __restrict__ h_bf) {
    long i = (long)blockIdx.x * blockDim.x + threadIdx.x;
    if (i >= (long)NCTX * HID) return;
    long n = i / HID;
    int  k = (int)(i % HID);
    const float* gr = g + n * FOURH;
    float gi = gr[k], gf = gr[HID + k], gg = gr[2 * HID + k], go = gr[3 * HID + k];
    float si = 1.0f / (1.0f + __expf(-gi));
    float sf = 1.0f / (1.0f + __expf(-gf));
    float so = 1.0f / (1.0f + __expf(-go));
    float cv = sf * c[i] + si * tanhf(gg);
    float hv = so * tanhf(cv);
    c[i] = cv;
    h_out[i] = hv;
    h_bf[i] = (bf16_t)hv;
}

// concat [ll | h_f | h_b | rl] and cast to bf16
__global__ void build_ctxin(const float* __restrict__ ll, const float* __restrict__ hf,
                            const float* __restrict__ hb, const float* __restrict__ rl,
                            bf16_t* __restrict__ out) {
    long i = (long)blockIdx.x * blockDim.x + threadIdx.x;
    if (i >= (long)NCTX * CTXW) return;
    long n = i / CTXW;
    int  col = (int)(i % CTXW);
    float v;
    if (col < EMB)                 v = ll[n * EMB + col];
    else if (col < EMB + HID)      v = hf[n * HID + (col - EMB)];
    else if (col < EMB + 2 * HID)  v = hb[n * HID + (col - EMB - HID)];
    else                           v = rl[n * EMB + (col - EMB - 2 * HID)];
    out[i] = (bf16_t)v;
}

// ctx = tanh(ctxin @ fc_w^T); grid = (NCTX/64, DEC/64); wave -> 16(M) x 64(N)
__global__ __launch_bounds__(128)
void fc_gemm_tanh(const bf16_t* __restrict__ A, const bf16_t* __restrict__ W,
                  float* __restrict__ ctx, bf16_t* __restrict__ ctx_bf) {
    const int lane = threadIdx.x & 31;
    const int wave = threadIdx.x >> 5;
    const int m0 = (blockIdx.x * 4 + wave) * 16;
    const int n0 = blockIdx.y * 64;
    const bf16_t* arow = A + (long)(m0 + (lane & 15)) * CTXW;
    const bf16_t* brow[4];
#pragma unroll
    for (int u = 0; u < 4; ++u)
        brow[u] = W + (long)(n0 + 16 * u + (lane & 15)) * CTXW;

    v8f acc[4] = {};
#pragma unroll 4
    for (int k0 = 0; k0 < CTXW; k0 += 32) {
        v16bf af = load_a_frag_row(arow, k0, lane);
#pragma unroll
        for (int u = 0; u < 4; ++u) {
            v16bf bf = load_b_frag_row(brow[u], k0, lane);
            acc[u] = __builtin_amdgcn_wmma_f32_16x16x32_bf16(false, af, false, bf,
                                                             (short)0, acc[u], false, false);
        }
    }
    const int mb = m0 + ((lane & 16) ? 8 : 0);
#pragma unroll
    for (int u = 0; u < 4; ++u) {
        const int n = n0 + 16 * u + (lane & 15);
#pragma unroll
        for (int r = 0; r < 8; ++r) {
            float tv = tanhf(acc[u][r]);
            long  o  = (long)(mb + r) * DEC + n;
            ctx[o]    = tv;
            ctx_bf[o] = (bf16_t)tv;
        }
    }
}

// score = ctx @ a; per-segment max via monotone-uint atomicMax (init 0 via memset)
__global__ void score_max(const float* __restrict__ ctx, const float* __restrict__ a,
                          const int* __restrict__ segs, float* __restrict__ score,
                          unsigned* __restrict__ segmax) {
    int n = blockIdx.x * blockDim.x + threadIdx.x;
    if (n >= NCTX) return;
    const float* r = ctx + (long)n * DEC;
    float s = 0.0f;
#pragma unroll 4
    for (int d = 0; d < DEC; ++d) s += r[d] * a[d];
    score[n] = s;
    unsigned u = __float_as_uint(s);
    unsigned key = (u & 0x80000000u) ? ~u : (u | 0x80000000u);
    atomicMax(&segmax[segs[n]], key);
}

__global__ void exp_sum(float* __restrict__ score, const unsigned* __restrict__ segmax,
                        const int* __restrict__ segs, float* __restrict__ segsum) {
    int n = blockIdx.x * blockDim.x + threadIdx.x;
    if (n >= NCTX) return;
    unsigned key = segmax[segs[n]];
    unsigned u = (key & 0x80000000u) ? (key ^ 0x80000000u) : ~key;
    float m = __uint_as_float(u);
    float e = __expf(score[n] - m);
    score[n] = e;
    atomicAdd(&segsum[segs[n]], e);
}

__global__ void attn_accum(const float* __restrict__ ctx, const float* __restrict__ e,
                           const float* __restrict__ segsum, const int* __restrict__ segs,
                           float* __restrict__ v) {
    long i = (long)blockIdx.x * blockDim.x + threadIdx.x;
    if (i >= (long)NCTX * DEC) return;
    long n = i / DEC;
    int  d = (int)(i % DEC);
    int sg = segs[n];
    float w = e[n] / segsum[sg];
    atomicAdd(&v[(long)sg * DEC + d], ctx[i] * w);
}

// out = v @ out_w^T + out_b; grid = (BSEG/256, OUTV/16); wave -> 64(M) x 16(N), B-frag reused 4x
__global__ __launch_bounds__(128)
void out_gemm(const bf16_t* __restrict__ A, const bf16_t* __restrict__ W,
              const float* __restrict__ bias, float* __restrict__ out) {
    const int lane = threadIdx.x & 31;
    const int wave = threadIdx.x >> 5;
    const int m0 = (blockIdx.x * 4 + wave) * 64;
    const int n0 = blockIdx.y * 16;
    const bf16_t* brow = W + (long)(n0 + (lane & 15)) * DEC;
    const bf16_t* arow[4];
#pragma unroll
    for (int s = 0; s < 4; ++s)
        arow[s] = A + (long)(m0 + 16 * s + (lane & 15)) * DEC;

    v8f acc[4] = {};
#pragma unroll
    for (int k0 = 0; k0 < DEC; k0 += 32) {
        v16bf bf = load_b_frag_row(brow, k0, lane);
#pragma unroll
        for (int s = 0; s < 4; ++s) {
            v16bf af = load_a_frag_row(arow[s], k0, lane);
            acc[s] = __builtin_amdgcn_wmma_f32_16x16x32_bf16(false, af, false, bf,
                                                             (short)0, acc[s], false, false);
        }
    }
    const int n  = n0 + (lane & 15);
    const float bv = bias[n];
#pragma unroll
    for (int s = 0; s < 4; ++s) {
        const int mb = m0 + 16 * s + ((lane & 16) ? 8 : 0);
#pragma unroll
        for (int r = 0; r < 8; ++r)
            out[(long)(mb + r) * OUTV + n] = acc[s][r] + bv;
    }
}

// ---------------- host launch ----------------
extern "C" void kernel_launch(void* const* d_in, const int* in_sizes, int n_in,
                              void* d_out, int out_size, void* d_ws, size_t ws_size,
                              hipStream_t stream) {
    const int*   ll_tok = (const int*)d_in[0];
    const int*   ll_idx = (const int*)d_in[1];
    const int*   rl_tok = (const int*)d_in[2];
    const int*   rl_idx = (const int*)d_in[3];
    const int*   paths  = (const int*)d_in[4];
    const int*   segs   = (const int*)d_in[5];
    const float* sub_emb  = (const float*)d_in[6];
    const float* node_emb = (const float*)d_in[7];
    const float* w_ih_f = (const float*)d_in[8];
    const float* w_hh_f = (const float*)d_in[9];
    const float* b_ih_f = (const float*)d_in[10];
    const float* b_hh_f = (const float*)d_in[11];
    const float* w_ih_b = (const float*)d_in[12];
    const float* w_hh_b = (const float*)d_in[13];
    const float* b_ih_b = (const float*)d_in[14];
    const float* b_hh_b = (const float*)d_in[15];
    const float* fc_w   = (const float*)d_in[16];
    const float* a_vec  = (const float*)d_in[17];
    const float* out_w  = (const float*)d_in[18];
    const float* out_b  = (const float*)d_in[19];
    float* out = (float*)d_out;
    const int L = in_sizes[0];

    // ---- bump-allocate workspace ----
    char* p = (char*)d_ws;
    auto alloc = [&](size_t bytes) -> void* {
        void* r = (void*)p;
        p += (bytes + 255) & ~(size_t)255;
        return r;
    };
    float*  ll_emb  = (float*)alloc((size_t)NCTX * EMB * 4);
    float*  rl_emb  = (float*)alloc((size_t)NCTX * EMB * 4);
    float*  h_f     = (float*)alloc((size_t)NCTX * HID * 4);
    float*  h_b     = (float*)alloc((size_t)NCTX * HID * 4);
    float*  c_buf   = (float*)alloc((size_t)NCTX * HID * 4);
    float*  g_buf   = (float*)alloc((size_t)NCTX * FOURH * 4);   // later aliased as ctx f32
    bf16_t* h_bf    = (bf16_t*)alloc((size_t)NCTX * HID * 2);
    bf16_t* ctxin   = (bf16_t*)alloc((size_t)NCTX * CTXW * 2);
    bf16_t* ctx_bf  = (bf16_t*)alloc((size_t)NCTX * DEC * 2);
    float*  score   = (float*)alloc((size_t)NCTX * 4);
    unsigned* segmx = (unsigned*)alloc((size_t)BSEG * 4);
    float*  segsum  = (float*)alloc((size_t)BSEG * 4);
    float*  v_buf   = (float*)alloc((size_t)BSEG * DEC * 4);
    bf16_t* v_bf    = (bf16_t*)alloc((size_t)BSEG * DEC * 2);
    bf16_t* wihf_bf = (bf16_t*)alloc((size_t)FOURH * EMB * 2);
    bf16_t* whhf_bf = (bf16_t*)alloc((size_t)FOURH * HID * 2);
    bf16_t* wihb_bf = (bf16_t*)alloc((size_t)FOURH * EMB * 2);
    bf16_t* whhb_bf = (bf16_t*)alloc((size_t)FOURH * HID * 2);
    bf16_t* fcw_bf  = (bf16_t*)alloc((size_t)DEC * CTXW * 2);
    bf16_t* outw_bf = (bf16_t*)alloc((size_t)OUTV * DEC * 2);
    bf16_t* node_bf = (bf16_t*)alloc((size_t)NODE_V * EMB * 2);
    float*  bias_f  = (float*)alloc((size_t)FOURH * 4);
    float*  bias_b  = (float*)alloc((size_t)FOURH * 4);
    float*  ctx_f32 = g_buf;   // alias: g dead after LSTM, DEC < FOURH

    // ---- zero-init accumulators ----
    hipMemsetAsync(ll_emb, 0, (size_t)NCTX * EMB * 4, stream);
    hipMemsetAsync(rl_emb, 0, (size_t)NCTX * EMB * 4, stream);
    hipMemsetAsync(segmx,  0, (size_t)BSEG * 4, stream);
    hipMemsetAsync(segsum, 0, (size_t)BSEG * 4, stream);
    hipMemsetAsync(v_buf,  0, (size_t)BSEG * DEC * 4, stream);

    // ---- cast weights to bf16 ----
    auto cast = [&](const float* s, bf16_t* d, long n) {
        cast_f32_to_bf16<<<dim3((unsigned)((n + 255) / 256)), dim3(256), 0, stream>>>(s, d, n);
    };
    cast(w_ih_f, wihf_bf, (long)FOURH * EMB);
    cast(w_hh_f, whhf_bf, (long)FOURH * HID);
    cast(w_ih_b, wihb_bf, (long)FOURH * EMB);
    cast(w_hh_b, whhb_bf, (long)FOURH * HID);
    cast(fc_w,   fcw_bf,  (long)DEC * CTXW);
    cast(out_w,  outw_bf, (long)OUTV * DEC);
    cast(node_emb, node_bf, (long)NODE_V * EMB);
    combine_bias<<<dim3(2), dim3(256), 0, stream>>>(b_ih_f, b_hh_f, bias_f, FOURH);
    combine_bias<<<dim3(2), dim3(256), 0, stream>>>(b_ih_b, b_hh_b, bias_b, FOURH);

    // ---- segment-sum of subtoken embeddings ----
    {
        long threads = (long)L * (EMB / 4);
        unsigned blocks = (unsigned)((threads + 255) / 256);
        seg_emb_add<<<dim3(blocks), dim3(256), 0, stream>>>(ll_tok, ll_idx, sub_emb, ll_emb, L);
        seg_emb_add<<<dim3(blocks), dim3(256), 0, stream>>>(rl_tok, rl_idx, sub_emb, rl_emb, L);
    }

    // ---- bidirectional LSTM ----
    const dim3 ggrid(NCTX / 64, FOURH / 64);
    const dim3 gblock(128);
    const unsigned gate_blocks = (unsigned)(((long)NCTX * HID + 255) / 256);
    // forward
    hipMemsetAsync(c_buf, 0, (size_t)NCTX * HID * 4, stream);
    hipMemsetAsync(h_bf,  0, (size_t)NCTX * HID * 2, stream);
    for (int t = 0; t < TLEN; ++t) {
        lstm_gemm_step<<<ggrid, gblock, 0, stream>>>(node_bf, paths, t, h_bf,
                                                     wihf_bf, whhf_bf, bias_f, g_buf);
        lstm_gate<<<dim3(gate_blocks), dim3(256), 0, stream>>>(g_buf, c_buf, h_f, h_bf);
    }
    // backward (reversed time)
    hipMemsetAsync(c_buf, 0, (size_t)NCTX * HID * 4, stream);
    hipMemsetAsync(h_bf,  0, (size_t)NCTX * HID * 2, stream);
    for (int t = TLEN - 1; t >= 0; --t) {
        lstm_gemm_step<<<ggrid, gblock, 0, stream>>>(node_bf, paths, t, h_bf,
                                                     wihb_bf, whhb_bf, bias_b, g_buf);
        lstm_gate<<<dim3(gate_blocks), dim3(256), 0, stream>>>(g_buf, c_buf, h_b, h_bf);
    }

    // ---- concat + FC(tanh) ----
    {
        unsigned blocks = (unsigned)(((long)NCTX * CTXW + 255) / 256);
        build_ctxin<<<dim3(blocks), dim3(256), 0, stream>>>(ll_emb, h_f, h_b, rl_emb, ctxin);
    }
    fc_gemm_tanh<<<dim3(NCTX / 64, DEC / 64), gblock, 0, stream>>>(ctxin, fcw_bf, ctx_f32, ctx_bf);

    // ---- segment softmax attention ----
    score_max<<<dim3(NCTX / 256), dim3(256), 0, stream>>>(ctx_f32, a_vec, segs, score, segmx);
    exp_sum<<<dim3(NCTX / 256), dim3(256), 0, stream>>>(score, segmx, segs, segsum);
    {
        unsigned blocks = (unsigned)(((long)NCTX * DEC + 255) / 256);
        attn_accum<<<dim3(blocks), dim3(256), 0, stream>>>(ctx_f32, score, segsum, segs, v_buf);
    }
    cast(v_buf, v_bf, (long)BSEG * DEC);

    // ---- output head ----
    out_gemm<<<dim3(BSEG / 256, OUTV / 16), gblock, 0, stream>>>(v_bf, outw_bf, out_b, out);
}
